// GCN_28630251995713
// MI455X (gfx1250) — matrix operand order repaired and checked by reading, not compiled, not run on hardware
//
#include <hip/hip_runtime.h>
#include <hip/hip_bf16.h>
#include <math.h>

// ---------------------------------------------------------------------------
// 4-layer GCN for MI455X (gfx1250, wave32).
// Dense GEMMs: bf16 WMMA (v_wmma_f32_16x16x32_bf16), f32 accumulate,
//   pre-packed B fragments, bit-cast A fragments, 4 row-tiles per block.
// Norm trick: feature buffers are stored pre-scaled by dinv, so the GEMM
//   writes raw accumulators and every edge is a pure atomic add.
// ---------------------------------------------------------------------------

typedef __attribute__((ext_vector_type(16))) __bf16         v16bf;
typedef __attribute__((ext_vector_type(8)))  float          v8f;
typedef __attribute__((ext_vector_type(4)))  float          v4f;
typedef __attribute__((ext_vector_type(4)))  unsigned int   v4u;
typedef __attribute__((ext_vector_type(4)))  unsigned short v4us;

#define FDIM 128

static __device__ __forceinline__ unsigned short f32_to_bf16(float f) {
  unsigned u = __float_as_uint(f);
  u += 0x7fffu + ((u >> 16) & 1u);       // round-to-nearest-even
  return (unsigned short)(u >> 16);
}
static __device__ __forceinline__ float bf16us_to_f32(unsigned short u) {
  return __uint_as_float(((unsigned)u) << 16);
}
static __device__ __forceinline__ void atomic_add_f32(float* p, float v) {
  __hip_atomic_fetch_add(p, v, __ATOMIC_RELAXED, __HIP_MEMORY_SCOPE_AGENT);
}

// -------------------------- degree / norm ----------------------------------
__global__ void gcn_degree_kernel(const int* __restrict__ dst, float* __restrict__ deg, int E) {
  int e = blockIdx.x * blockDim.x + threadIdx.x;
  if (e < E) atomic_add_f32(&deg[dst[e]], 1.0f);
}

__global__ void gcn_dinv_kernel(float* __restrict__ deg_dinv, int N) {
  int i = blockIdx.x * blockDim.x + threadIdx.x;
  if (i < N) deg_dinv[i] = rsqrtf(deg_dinv[i] + 1.0f);  // +1 = self loop
}

// ---------- f32 -> bf16 convert with per-row dinv scale (x4 vector) --------
// dst[i][f] = bf16( dinv[i] * src[i][f] )
__global__ void gcn_cvt_scale_bf16x4_kernel(const float* __restrict__ src,
                                            const float* __restrict__ dinv,
                                            unsigned short* __restrict__ dst, int n4) {
  int q = blockIdx.x * blockDim.x + threadIdx.x;
  if (q >= n4) return;
  const float di = dinv[q >> 5];           // 32 quads per 128-feature row
  v4f s = *(const v4f*)(src + 4 * (size_t)q);
  v4us o;
  o.x = f32_to_bf16(s.x * di); o.y = f32_to_bf16(s.y * di);
  o.z = f32_to_bf16(s.z * di); o.w = f32_to_bf16(s.w * di);
  *(v4us*)(dst + 4 * (size_t)q) = o;
}

// -------------------- pack W (f32 row-major) into B-fragment order ---------
// Wpack[((t*4 + kt)*32 + lane)*16 + j] = bf16( W[(kt*32 + (lane>>4)*16 + j)*128
//                                              + t*16 + (lane&15)] )
__global__ void gcn_pack_w_kernel(const float* __restrict__ W,
                                  unsigned short* __restrict__ Wpack) {
  int idx = blockIdx.x * blockDim.x + threadIdx.x;   // 0 .. 16383
  if (idx >= FDIM * FDIM) return;
  const int j    = idx & 15;
  const int lane = (idx >> 4) & 31;
  const int kt   = (idx >> 9) & 3;
  const int t    = idx >> 11;
  const int k    = kt * 32 + (lane >> 4) * 16 + j;
  const int col  = t * 16 + (lane & 15);
  Wpack[idx] = f32_to_bf16(W[(size_t)k * FDIM + col]);
}

// ------------------------------ WMMA GEMM ----------------------------------
// hs = A @ W.  A: nrows x 128 bf16 row-major (already dinv-scaled),
// Wpack: pre-packed B fragments.  Block = 8 waves x 4 row tiles (64 rows).
__global__ void gcn_gemm_wmma_kernel(const unsigned short* __restrict__ Abf,
                                     const unsigned short* __restrict__ Wpack,
                                     float* __restrict__ hs, int nrows) {
  const int wave = threadIdx.x >> 5;       // 0..7 -> output column tile
  const int lane = threadIdx.x & 31;
  const int half = lane >> 4;              // lane half selects K sub-range
  const int l15  = lane & 15;
  const int col0 = wave * 16;

  // B fragments for this wave's column tile: resident across all row tiles.
  const v16bf* Bp = (const v16bf*)Wpack;
  v16bf bfrag[4];
  #pragma unroll
  for (int kt = 0; kt < 4; ++kt)
    bfrag[kt] = Bp[(size_t)(wave * 4 + kt) * 32 + lane];

  union AFrag { v16bf v; v4u q[2]; };

  #pragma unroll
  for (int mt = 0; mt < 4; ++mt) {
    const int row0 = blockIdx.x * 64 + mt * 16;
    if (row0 >= nrows) break;              // uniform across wave

    int arow = row0 + l15;
    if (arow >= nrows) arow = nrows - 1;   // clamp loads (tail tile only)
    const v4u* ap = (const v4u*)(Abf + (size_t)arow * FDIM);  // 16B groups

    v8f acc = {};
    #pragma unroll
    for (int kt = 0; kt < 4; ++kt) {
      // A fragment 16x32 bf16, contiguous K order matching Wpack:
      //   u[0..3] = K half*16 .. half*16+7 of the K-tile
      //   u[4..7] = K half*16+8 .. half*16+15
      AFrag a;
      a.q[0] = ap[kt * 4 + half * 2];
      a.q[1] = ap[kt * 4 + half * 2 + 1];
      acc = __builtin_amdgcn_wmma_f32_16x16x32_bf16(
          /*neg_a=*/false, a.v, /*neg_b=*/false, bfrag[kt],
          /*c_mod=*/(short)0, acc, /*reuse_a=*/false, /*reuse_b=*/false);
    }

    // D layout: VGPR r -> M = r + 8*half, N = lane&15.
    float* out = hs + (size_t)(row0 + half * 8) * FDIM + col0 + l15;
    if (row0 + 16 <= nrows) {              // uniform: full tile, no guards
      #pragma unroll
      for (int r = 0; r < 8; ++r)
        out[(size_t)r * FDIM] = acc[r];
    } else {                               // tail tile only
      #pragma unroll
      for (int r = 0; r < 8; ++r)
        if (row0 + half * 8 + r < nrows)
          out[(size_t)r * FDIM] = acc[r];
    }
  }
}

// -------------------------- edge scatter (128 feats) -----------------------
// One 32-lane group per edge: agg[dst] += hs[src]  (norm already folded in).
__global__ void gcn_scatter_kernel(const int* __restrict__ src,
                                   const int* __restrict__ dst,
                                   const float* __restrict__ hs,
                                   float* __restrict__ agg, int E) {
  const int g    = (blockIdx.x * blockDim.x + threadIdx.x) >> 5;
  const int lane = threadIdx.x & 31;
  if (g >= E) return;
  const int s = src[g], d = dst[g];
  const float* hp = hs  + (size_t)s * FDIM;
  float*       ap = agg + (size_t)d * FDIM;
  #pragma unroll
  for (int f = 0; f < FDIM; f += 32)
    atomic_add_f32(ap + f + lane, hp[f + lane]);
}

// -------------------------- layer epilogue (x4 vectorized) -----------------
// H = relu((agg + hs_self) * dinv + b); store H' = dinv * H as bf16.
__global__ void gcn_epilogue_kernel(const float* __restrict__ agg,
                                    const float* __restrict__ hs,
                                    const float* __restrict__ dinv,
                                    const float* __restrict__ bias,
                                    unsigned short* __restrict__ out_bf, int N) {
  const int q = blockIdx.x * blockDim.x + threadIdx.x;   // N*128/4 quads
  if (q >= N * (FDIM / 4)) return;
  const size_t idx = (size_t)q * 4;
  const int i = q >> 5;                    // node
  const int f = (q & 31) * 4;              // feature base
  const float di = dinv[i];
  const v4f va = *(const v4f*)(agg + idx);
  const v4f vh = *(const v4f*)(hs + idx);
  const v4f vb = *(const v4f*)(bias + f);
  v4us o;
  o.x = f32_to_bf16(fmaxf((va.x + vh.x) * di + vb.x, 0.0f) * di);
  o.y = f32_to_bf16(fmaxf((va.y + vh.y) * di + vb.y, 0.0f) * di);
  o.z = f32_to_bf16(fmaxf((va.z + vh.z) * di + vb.z, 0.0f) * di);
  o.w = f32_to_bf16(fmaxf((va.w + vh.w) * di + vb.w, 0.0f) * di);
  *(v4us*)(out_bf + idx) = o;
}

// -------------------------- final 128 -> 2 layer ---------------------------
// Input already dinv-scaled, so hs4 = H'_3 @ W4 directly.
__global__ void gcn_final_linear_kernel(const unsigned short* __restrict__ in_bf,
                                        const float* __restrict__ W4,
                                        float* __restrict__ hs4, int N) {
  const int i = blockIdx.x * blockDim.x + threadIdx.x;
  if (i >= N) return;
  const unsigned short* p = in_bf + (size_t)i * FDIM;
  float a0 = 0.f, a1 = 0.f;
  #pragma unroll 8
  for (int k = 0; k < FDIM; ++k) {
    const float xv = bf16us_to_f32(p[k]);
    a0 += xv * W4[2 * k];
    a1 += xv * W4[2 * k + 1];
  }
  hs4[2 * i]     = a0;
  hs4[2 * i + 1] = a1;
}

__global__ void gcn_scatter2_kernel(const int* __restrict__ src,
                                    const int* __restrict__ dst,
                                    const float* __restrict__ hs4,
                                    float* __restrict__ agg2, int E) {
  const int e = blockIdx.x * blockDim.x + threadIdx.x;
  if (e >= E) return;
  const int s = src[e], d = dst[e];
  atomic_add_f32(&agg2[2 * d],     hs4[2 * s]);
  atomic_add_f32(&agg2[2 * d + 1], hs4[2 * s + 1]);
}

__global__ void gcn_logsoftmax_kernel(const float* __restrict__ agg2,
                                      const float* __restrict__ hs4,
                                      const float* __restrict__ dinv,
                                      const float* __restrict__ b4,
                                      float* __restrict__ out, int N) {
  const int i = blockIdx.x * blockDim.x + threadIdx.x;
  if (i >= N) return;
  const float di = dinv[i];
  const float z0 = (agg2[2 * i]     + hs4[2 * i])     * di + b4[0];
  const float z1 = (agg2[2 * i + 1] + hs4[2 * i + 1]) * di + b4[1];
  const float m   = fmaxf(z0, z1);
  const float lse = m + logf(expf(z0 - m) + expf(z1 - m));
  out[2 * i]     = z0 - lse;
  out[2 * i + 1] = z1 - lse;
}

// ---------------------------------------------------------------------------
extern "C" void kernel_launch(void* const* d_in, const int* in_sizes, int n_in,
                              void* d_out, int out_size, void* d_ws, size_t ws_size,
                              hipStream_t stream) {
  const float* x    = (const float*)d_in[0];
  const int*   ei   = (const int*)d_in[1];          // [2][E] (int32)
  const float* W[3] = { (const float*)d_in[2], (const float*)d_in[4],
                        (const float*)d_in[6] };
  const float* B[3] = { (const float*)d_in[3], (const float*)d_in[5],
                        (const float*)d_in[7] };
  const float* W4   = (const float*)d_in[8];
  const float* b4   = (const float*)d_in[9];

  const int N = in_sizes[0] / FDIM;
  const int E = in_sizes[1] / 2;
  const int* esrc = ei;
  const int* edst = ei + E;

  // ---- workspace carve-out (256B aligned) ----
  char* w = (char*)d_ws;
  auto carve = [&](size_t bytes) {
    void* p = (void*)w;
    w += (bytes + 255) & ~(size_t)255;
    return p;
  };
  float*          dinv = (float*)carve((size_t)N * sizeof(float));       // deg -> dinv
  float*          hs   = (float*)carve((size_t)N * FDIM * sizeof(float));
  float*          agg  = (float*)carve((size_t)N * FDIM * sizeof(float));
  float*          hs4  = (float*)carve((size_t)N * 2 * sizeof(float));
  float*          agg2 = (float*)carve((size_t)N * 2 * sizeof(float));
  unsigned short* fb0  = (unsigned short*)carve((size_t)N * FDIM * 2);   // ping
  unsigned short* fb1  = (unsigned short*)carve((size_t)N * FDIM * 2);   // pong
  unsigned short* Wpk[3];
  for (int l = 0; l < 3; ++l)
    Wpk[l] = (unsigned short*)carve((size_t)FDIM * FDIM * 2);

  const int TB = 256;
  auto nb = [&](long long n) { return (unsigned)((n + TB - 1) / TB); };

  // ---- degree / dinv ----
  hipMemsetAsync(dinv, 0, (size_t)N * sizeof(float), stream);
  gcn_degree_kernel<<<nb(E), TB, 0, stream>>>(edst, dinv, E);
  gcn_dinv_kernel<<<nb(N), TB, 0, stream>>>(dinv, N);

  // ---- dinv-scaled bf16 conversion of x, fragment-packing of W1..W3 ----
  gcn_cvt_scale_bf16x4_kernel<<<nb((long long)N * (FDIM / 4)), TB, 0, stream>>>(
      x, dinv, fb0, N * (FDIM / 4));
  for (int l = 0; l < 3; ++l)
    gcn_pack_w_kernel<<<nb(FDIM * FDIM), TB, 0, stream>>>(W[l], Wpk[l]);

  // ---- three hidden GCN layers ----
  unsigned short* cur = fb0;
  unsigned short* nxt = fb1;
  const unsigned gemm_blocks = (unsigned)((N + 63) / 64);   // 64 rows / block
  for (int l = 0; l < 3; ++l) {
    gcn_gemm_wmma_kernel<<<gemm_blocks, TB, 0, stream>>>(cur, Wpk[l], hs, N);
    hipMemsetAsync(agg, 0, (size_t)N * FDIM * sizeof(float), stream);
    gcn_scatter_kernel<<<nb((long long)E * 32), TB, 0, stream>>>(esrc, edst, hs, agg, E);
    gcn_epilogue_kernel<<<nb((long long)N * (FDIM / 4)), TB, 0, stream>>>(
        agg, hs, dinv, B[l], nxt, N);
    unsigned short* t = cur; cur = nxt; nxt = t;
  }

  // ---- final 128 -> 2 layer + log_softmax ----
  gcn_final_linear_kernel<<<nb(N), TB, 0, stream>>>(cur, W4, hs4, N);
  hipMemsetAsync(agg2, 0, (size_t)N * 2 * sizeof(float), stream);
  gcn_scatter2_kernel<<<nb(E), TB, 0, stream>>>(esrc, edst, hs4, agg2, E);
  gcn_logsoftmax_kernel<<<nb(N), TB, 0, stream>>>(agg2, hs4, dinv, b4, (float*)d_out, N);
}
